// SymNCC_28226525070036
// MI455X (gfx1250) — matrix operand order repaired and checked by reading, not compiled, not run on hardware
//
#include <hip/hip_runtime.h>

// ---------------------------------------------------------------------------
// Fused LNCC (win=5, eps=1e-5) for (2,1,192,192,192) f32 volumes on gfx1250.
// Single pass over HBM; x-filter on VALU from LDS, y-filter on the matrix
// pipe (v_wmma_f32_16x16x32_f16, banded 0/1 matrix), z via f32 sliding-window
// accumulator with an LDS ring. Interior tiles stream raw data with
// global_load_async_to_lds_b32 (ASYNCcnt path).
// ---------------------------------------------------------------------------

typedef __attribute__((ext_vector_type(16))) _Float16 v16h;
typedef __attribute__((ext_vector_type(2)))  _Float16 h2;
typedef __attribute__((ext_vector_type(8)))  float    v8f;

#define DIM     192
#define NBATCH  2
#define TILE_X  16
#define WAVES   2
#define TILE_Y  (16 * WAVES)          // 32
#define ZC      24
#define NCHUNK  (DIM / ZC)            // 8
#define RAWY    (TILE_Y + 4)          // 36
#define RAWX    (TILE_X + 4)          // 20
#define GX      (DIM / TILE_X)        // 12
#define GY      (DIM / TILE_Y)        // 6
#define GZ      (NBATCH * NCHUNK)     // 16
#define NBLOCKS (GX * GY * GZ)        // 1152
#define TROW    24                    // tmpT row length in halves (k-pad to 24)
#define WIN_INV (1.0f / 125.0f)
#define EPS_T   1e-5f

__global__ __launch_bounds__(64) void lncc_main(const float* __restrict__ I,
                                                const float* __restrict__ J,
                                                float* __restrict__ partial) {
    __shared__ float rawI[RAWY][RAWX];
    __shared__ float rawJ[RAWY][RAWX];
    // x-filtered quantities, transposed: row = (w*5+q)*16 + n, element row*TROW+k.
    // k=20..23 zero pad; +8-half zero tail so the last overrun read is finite.
    __shared__ __align__(64) _Float16 tmpT[WAVES * 5 * 16 * TROW + 8];
    // z ring of 2D-filtered planes (f32: exact sliding-window cancellation)
    __shared__ float ring[5][WAVES][5][8][32];

    const int tid  = threadIdx.x;
    const int lane = tid & 31;
    const int w    = tid >> 5;                  // wave id (0..1)
    const int n    = lane & 15;                 // N (x) index for WMMA B/D
    const int x0   = blockIdx.x * TILE_X;
    const int y0   = blockIdx.y * TILE_Y;
    const int b    = blockIdx.z / NCHUNK;
    const int z0   = (blockIdx.z % NCHUNK) * ZC;

    // x/y-interior blocks never touch the zero-padded border (uniform predicate)
    const bool xyint = (blockIdx.x >= 1) & (blockIdx.x <= GX - 2) &
                       (blockIdx.y >= 1) & (blockIdx.y <= GY - 2);

    // Constant banded A fragment: A[m][k] = 1 iff m <= k <= m+4 (k < 20).
    v16h afrag;
    {
        const int m = lane & 15;
#pragma unroll
        for (int h = 0; h < 16; ++h) {
            int k = (lane & 16) + h;
            afrag[h] = (_Float16)((k < 20 && k >= m && k <= m + 4) ? 1.0f : 0.0f);
        }
    }

    // One-time zeroing: tmpT pads (k=20..23 per row, 8-half tail) and the ring.
    for (int r = tid; r < WAVES * 5 * 16; r += 64) {
        *(h2*)&tmpT[r * TROW + 20] = (h2){};
        *(h2*)&tmpT[r * TROW + 22] = (h2){};
    }
    if (tid < 4) *(h2*)&tmpT[WAVES * 5 * 16 * TROW + 2 * tid] = (h2){};
    {
        float* rp = &ring[0][0][0][0][0];
        for (int i = tid; i < 5 * WAVES * 5 * 8 * 32; i += 64) rp[i] = 0.0f;
    }

    v8f acc[5];
#pragma unroll
    for (int q = 0; q < 5; ++q) acc[q] = (v8f){};

    float ccsum = 0.0f;
    const int planeBase = b * DIM * DIM * DIM;

    __syncthreads();

    for (int zp = z0 - 2; zp <= z0 + ZC + 1; ++zp) {
        // ---- A: raw halo plane -> LDS ---------------------------------------
        const bool zin  = (zp >= 0) & (zp < DIM);
        const int zbase = planeBase + zp * DIM * DIM;
        if (xyint & zin) {
            // fully in-bounds: async DMA straight into LDS (ASYNCcnt path)
#pragma unroll
            for (int itl = 0; itl < 12; ++itl) {
                int i = itl * 64 + tid;
                if (i < RAWY * RAWX) {
                    int ry = i / RAWX, rx = i - ry * RAWX;
                    int gy = y0 - 2 + ry, gx = x0 - 2 + rx;
                    int idx = zbase + gy * DIM + gx;
                    unsigned li = (unsigned)(uintptr_t)&rawI[ry][rx];
                    unsigned lj = (unsigned)(uintptr_t)&rawJ[ry][rx];
                    unsigned long long ga = (unsigned long long)(uintptr_t)(I + idx);
                    unsigned long long gb = (unsigned long long)(uintptr_t)(J + idx);
                    asm volatile(
                        "global_load_async_to_lds_b32 %0, %1, off\n\t"
                        "global_load_async_to_lds_b32 %2, %3, off"
                        :: "v"(li), "v"(ga), "v"(lj), "v"(gb) : "memory");
                }
            }
            asm volatile("s_wait_asynccnt 0x0" ::: "memory");
        } else {
            // border: scalar loads with zero fill
#pragma unroll
            for (int itl = 0; itl < 12; ++itl) {
                int i = itl * 64 + tid;
                if (i < RAWY * RAWX) {
                    int ry = i / RAWX, rx = i - ry * RAWX;
                    int gy = y0 - 2 + ry, gx = x0 - 2 + rx;
                    float vi = 0.0f, vj = 0.0f;
                    if (zin & (gy >= 0) & (gy < DIM) & (gx >= 0) & (gx < DIM)) {
                        int idx = zbase + gy * DIM + gx;
                        vi = I[idx];
                        vj = J[idx];
                    }
                    rawI[ry][rx] = vi;
                    rawJ[ry][rx] = vj;
                }
            }
        }
        __syncthreads();

        // ---- B: x-direction 5-tap filter of 5 quantities -> tmpT (f16) -----
#pragma unroll
        for (int itb = 0; itb < 5; ++itb) {
            int pos = itb * 32 + lane;
            int xx  = pos & 15;
            int r   = (pos >> 4) * 2;           // 0,2,..,18
            int rr  = w * 16 + r;
            float sI0 = 0.f, sJ0 = 0.f, sI20 = 0.f, sJ20 = 0.f, sIJ0 = 0.f;
            float sI1 = 0.f, sJ1 = 0.f, sI21 = 0.f, sJ21 = 0.f, sIJ1 = 0.f;
#pragma unroll
            for (int d = 0; d < 5; ++d) {
                float a0 = rawI[rr][xx + d],     b0 = rawJ[rr][xx + d];
                float a1 = rawI[rr + 1][xx + d], b1 = rawJ[rr + 1][xx + d];
                sI0 += a0; sJ0 += b0; sI20 += a0 * a0; sJ20 += b0 * b0; sIJ0 += a0 * b0;
                sI1 += a1; sJ1 += b1; sI21 += a1 * a1; sJ21 += b1 * b1; sIJ1 += a1 * b1;
            }
            const int rb = (w * 5) * 16 + xx;   // row of q=0
            *(h2*)&tmpT[(rb +  0) * TROW + r] = (h2){(_Float16)sI0,  (_Float16)sI1};
            *(h2*)&tmpT[(rb + 16) * TROW + r] = (h2){(_Float16)sJ0,  (_Float16)sJ1};
            *(h2*)&tmpT[(rb + 32) * TROW + r] = (h2){(_Float16)sI20, (_Float16)sI21};
            *(h2*)&tmpT[(rb + 48) * TROW + r] = (h2){(_Float16)sJ20, (_Float16)sJ21};
            *(h2*)&tmpT[(rb + 64) * TROW + r] = (h2){(_Float16)sIJ0, (_Float16)sIJ1};
        }
        __syncthreads();

        // ---- C: y-filter on the matrix pipe + z sliding window -------------
        const int slot  = (zp + 10) % 5;
        const int kbase = lane & 16;            // B layout: K = (lane<16? h : 16+h)
#pragma unroll
        for (int q = 0; q < 5; ++q) {
            const int row = (w * 5 + q) * 16 + n;
            v16h bfrag = *(const v16h*)&tmpT[row * TROW + kbase];
            v8f zero = {};
            v8f f = __builtin_amdgcn_wmma_f32_16x16x32_f16(
                false, afrag, false, bfrag, (short)0, zero, false, false);
#pragma unroll
            for (int r = 0; r < 8; ++r) {
                float oldv = ring[slot][w][q][r][lane];
                acc[q][r] += f[r] - oldv;       // exact cancellation 5 planes later
                ring[slot][w][q][r][lane] = f[r];
            }
        }

        // ---- D: emit cc for z = zp-2 once the 5-deep z window is full ------
        if (zp >= z0 + 2) {
#pragma unroll
            for (int r = 0; r < 8; ++r) {
                float Is  = acc[0][r];
                float Js  = acc[1][r];
                float I2s = acc[2][r];
                float J2s = acc[3][r];
                float IJs = acc[4][r];
                float cross = IJs - Is * Js * WIN_INV;
                float Iv    = I2s - Is * Is * WIN_INV;
                float Jv    = J2s - Js * Js * WIN_INV;
                ccsum += (cross * cross + EPS_T) / (Iv * Jv + EPS_T);
            }
        }
    }

    // ---- block reduction (reuse rawI as scratch) ---------------------------
    __syncthreads();
    float* red = &rawI[0][0];
    red[tid] = ccsum;
    __syncthreads();
#pragma unroll
    for (int ofs = 32; ofs > 0; ofs >>= 1) {
        if (tid < ofs) red[tid] += red[tid + ofs];
        __syncthreads();
    }
    if (tid == 0) {
        int bid = (blockIdx.z * gridDim.y + blockIdx.y) * gridDim.x + blockIdx.x;
        partial[bid] = red[0];
    }
}

// Deterministic final reduction (fixed order, no float atomics).
__global__ __launch_bounds__(256) void lncc_reduce(const float* __restrict__ partial,
                                                   float* __restrict__ out) {
    __shared__ float buf[256];
    float s = 0.0f;
    for (int i = threadIdx.x; i < NBLOCKS; i += 256) s += partial[i];
    buf[threadIdx.x] = s;
    __syncthreads();
    for (int ofs = 128; ofs > 0; ofs >>= 1) {
        if (threadIdx.x < ofs) buf[threadIdx.x] += buf[threadIdx.x + ofs];
        __syncthreads();
    }
    if (threadIdx.x == 0)
        out[0] = buf[0] * (1.0f / ((float)NBATCH * DIM * DIM * DIM));
}

extern "C" void kernel_launch(void* const* d_in, const int* in_sizes, int n_in,
                              void* d_out, int out_size, void* d_ws, size_t ws_size,
                              hipStream_t stream) {
    const float* I = (const float*)d_in[0];
    const float* J = (const float*)d_in[1];
    float* partial = (float*)d_ws;

    dim3 grid(GX, GY, GZ);
    lncc_main<<<grid, dim3(64), 0, stream>>>(I, J, partial);
    lncc_reduce<<<1, 256, 0, stream>>>(partial, (float*)d_out);
}
